// GIN_7713761263896
// MI455X (gfx1250) — matrix operand order repaired and checked by reading, not compiled, not run on hardware
//
#include <hip/hip_runtime.h>
#include <cstdint>
#include <cstddef>

#define CN 100000   // nodes
#define CE 250000   // edges
#define CG 4096     // graphs
#define CD 300      // hidden dim
#define CL 5        // layers
#define AV 129      // padded atom vocab
#define BVV 16      // padded bond vocab
#define KP2 160     // packed K pair-rows (padded K = 320)
#define PW 320      // padded N width of packed weights
#define CSTRIDE 164 // LDS B-panel column stride (dwords): 36 mod 64, conflict-free tiling

typedef __attribute__((ext_vector_type(16))) __bf16 v16bf;
typedef __attribute__((ext_vector_type(8)))  float  v8f;

union BFrag { v16bf v; unsigned int u[8]; uint4 q[2]; };

__device__ __forceinline__ unsigned int f2bf_bits(float f) {
  union { float f; unsigned int u; } cv; cv.f = f;
  unsigned int x = cv.u;
  return (x + 0x7FFFu + ((x >> 16) & 1u)) >> 16;   // round-to-nearest-even
}
__device__ __forceinline__ unsigned int pack_bf(float lo, float hi) {
  return f2bf_bits(lo) | (f2bf_bits(hi) << 16);
}

// ---------------------------------------------------------------------------
// Weight packing: W[300x300] f32 row-major (K x N) -> Wp[160][320] dwords,
// each dword = bf16 pair (k, k+1) for column n. Zero-padded.
// ---------------------------------------------------------------------------
__global__ void pack_weights(const float* __restrict__ W, unsigned int* __restrict__ Wp,
                             int nmats) {
  int gid = blockIdx.x * blockDim.x + threadIdx.x;
  int per = KP2 * PW;
  if (gid >= nmats * per) return;
  int mat = gid / per;
  int rem = gid - mat * per;
  int kk = rem / PW;
  int n  = rem - kk * PW;
  int k0 = 2 * kk;
  const float* Wm = W + (size_t)mat * CD * CD;
  unsigned int lo = 0, hi = 0;
  if (n < CD) {
    if (k0     < CD) lo = f2bf_bits(Wm[(size_t)k0 * CD + n]);
    if (k0 + 1 < CD) hi = f2bf_bits(Wm[(size_t)(k0 + 1) * CD + n]);
  }
  Wp[gid] = lo | (hi << 16);
}

// ---------------------------------------------------------------------------
// WMMA GEMM: Y[M,300] = unpack(Xp) @ W + bias. Xp = [M][160] packed bf16 pairs.
// 8-wave block: 128 rows x 64 cols. B panel async-staged to LDS column-major
// so each B fragment is two ds_load_b128s; A fragment is two global b128s.
// ---------------------------------------------------------------------------
__global__ __launch_bounds__(256) void gemm_bf16_wmma(
    const unsigned int* __restrict__ Xp, const unsigned int* __restrict__ Wp,
    const float* __restrict__ bias, float* __restrict__ Y, int M) {
  __shared__ unsigned int BsT[64 * CSTRIDE];   // [col][kk], 41 KB
  const int tid  = threadIdx.x;
  const int lane = tid & 31;
  const int wave = tid >> 5;
  const int lm   = lane & 15;
  const int hi   = lane >> 4;
  const int colbase = blockIdx.x * 64;
  const int row0 = blockIdx.y * 128 + wave * 16;
  const bool rowOK = (row0 < M);

  // --- async-copy the 160x64 dword B panel into LDS, transposed ---
  // consecutive lanes read consecutive global columns (coalesced) and
  // scatter into BsT[c][kk].
  for (int i = tid; i < KP2 * 64; i += 256) {
    int kk = i >> 6;
    int c  = i & 63;
    unsigned lds_off = (unsigned)(size_t)&BsT[c * CSTRIDE + kk];
    unsigned long long gaddr =
        (unsigned long long)(size_t)(Wp + (size_t)kk * PW + colbase + c);
    asm volatile("global_load_async_to_lds_b32 %0, %1, off"
                 :: "v"(lds_off), "v"(gaddr) : "memory");
  }
  asm volatile("s_wait_asynccnt 0x0" ::: "memory");
  __syncthreads();

  const uint4* Arow =
      (const uint4*)(Xp + (size_t)(rowOK ? (row0 + lm) : lm) * KP2);

  v8f acc0 = {}, acc1 = {}, acc2 = {}, acc3 = {};

  for (int kt = 0; kt < 10; ++kt) {
    // A fragment: two contiguous b128 loads (pair-index layout)
    BFrag a;
    a.q[0] = Arow[kt * 4 + 2 * hi];
    a.q[1] = Arow[kt * 4 + 2 * hi + 1];
    // B fragments: 8 consecutive pair-rows per lane -> 2 x ds_load_b128 each
    const int kbase = kt * 16 + (hi ? 8 : 0);
    BFrag b0, b1, b2, b3;
    {
      const uint4* bp0 = (const uint4*)&BsT[(0 * 16 + lm) * CSTRIDE + kbase];
      const uint4* bp1 = (const uint4*)&BsT[(1 * 16 + lm) * CSTRIDE + kbase];
      const uint4* bp2 = (const uint4*)&BsT[(2 * 16 + lm) * CSTRIDE + kbase];
      const uint4* bp3 = (const uint4*)&BsT[(3 * 16 + lm) * CSTRIDE + kbase];
      b0.q[0] = bp0[0]; b0.q[1] = bp0[1];
      b1.q[0] = bp1[0]; b1.q[1] = bp1[1];
      b2.q[0] = bp2[0]; b2.q[1] = bp2[1];
      b3.q[0] = bp3[0]; b3.q[1] = bp3[1];
    }
    acc0 = __builtin_amdgcn_wmma_f32_16x16x32_bf16(false, a.v, false, b0.v, (short)0, acc0, false, false);
    acc1 = __builtin_amdgcn_wmma_f32_16x16x32_bf16(false, a.v, false, b1.v, (short)0, acc1, false, false);
    acc2 = __builtin_amdgcn_wmma_f32_16x16x32_bf16(false, a.v, false, b2.v, (short)0, acc2, false, false);
    acc3 = __builtin_amdgcn_wmma_f32_16x16x32_bf16(false, a.v, false, b3.v, (short)0, acc3, false, false);
  }

#define STORE_TILE(ACC, NT)                                                    \
  do {                                                                         \
    int col = colbase + (NT)*16 + lm;                                          \
    if (rowOK && col < CD) {                                                   \
      float bsv = bias[col];                                                   \
      float* yb = Y + (size_t)(row0 + hi * 8) * CD + col;                      \
      _Pragma("unroll")                                                        \
      for (int v = 0; v < 8; ++v) yb[(size_t)v * CD] = (ACC)[v] + bsv;         \
    }                                                                          \
  } while (0)

  STORE_TILE(acc0, 0);
  STORE_TILE(acc1, 1);
  STORE_TILE(acc2, 2);
  STORE_TILE(acc3, 3);
#undef STORE_TILE
}

// ---------------------------------------------------------------------------
// Elementwise / graph kernels
// ---------------------------------------------------------------------------
__global__ void fill_f32(float* __restrict__ p, float v, long n) {
  long gid = (long)blockIdx.x * blockDim.x + threadIdx.x;
  if (gid < n) p[gid] = v;
}

__global__ void atom_embed(const int* __restrict__ x, const float* __restrict__ emb,
                           float* __restrict__ h) {
  int gid = blockIdx.x * blockDim.x + threadIdx.x;
  if (gid >= CN * CD) return;
  int n = gid / CD, c = gid - n * CD;
  float s = 0.f;
#pragma unroll
  for (int i = 0; i < 9; ++i) {
    int idx = x[n * 9 + i];
    s += emb[((size_t)i * AV + idx) * CD + c];
  }
  h[gid] = s;
}

__global__ void bond_embed(const int* __restrict__ eattr, const float* __restrict__ emb,
                           float* __restrict__ ea) {
  long gid = (long)blockIdx.x * blockDim.x + threadIdx.x;
  if (gid >= (long)CE * CD) return;
  int e = (int)(gid / CD), c = (int)(gid - (long)e * CD);
  float s = 0.f;
#pragma unroll
  for (int i = 0; i < 3; ++i) {
    int idx = eattr[e * 3 + i];
    s += emb[((size_t)i * BVV + idx) * CD + c];
  }
  ea[gid] = s;
}

__global__ void broadcast_vn(const float* __restrict__ vne, float* __restrict__ vn) {
  int gid = blockIdx.x * blockDim.x + threadIdx.x;
  if (gid >= CG * CD) return;
  vn[gid] = vne[gid % CD];
}

__global__ void add_vn(float* __restrict__ xcur, const float* __restrict__ vn,
                       const int* __restrict__ batch) {
  int gid = blockIdx.x * blockDim.x + threadIdx.x;
  if (gid >= CN * CD) return;
  int n = gid / CD, c = gid - n * CD;
  xcur[gid] += vn[(size_t)batch[n] * CD + c];
}

__global__ void edge_scatter(const float* __restrict__ xcur, const float* __restrict__ ea,
                             const int* __restrict__ eidx, float* __restrict__ agg) {
  int gid = blockIdx.x * blockDim.x + threadIdx.x;
  const int total = CE * (CD / 4);
  if (gid >= total) return;
  int e = gid / (CD / 4);
  int c = (gid - e * (CD / 4)) * 4;
  int s = eidx[e];
  int d = eidx[CE + e];
  float4 xv = *(const float4*)(xcur + (size_t)s * CD + c);
  float4 ev = *(const float4*)(ea + (size_t)e * CD + c);
  float4 m;
  m.x = fmaxf(xv.x + ev.x, 0.f);
  m.y = fmaxf(xv.y + ev.y, 0.f);
  m.z = fmaxf(xv.z + ev.z, 0.f);
  m.w = fmaxf(xv.w + ev.w, 0.f);
  float* dp = agg + (size_t)d * CD + c;
  atomicAdd(dp + 0, m.x);
  atomicAdd(dp + 1, m.y);
  atomicAdd(dp + 2, m.z);
  atomicAdd(dp + 3, m.w);
}

// t0 = A + B, written directly as packed bf16 pairs [M][160]
__global__ void add_xy_pack(const float* __restrict__ A, const float* __restrict__ B,
                            unsigned int* __restrict__ Xp, int M) {
  int gid = blockIdx.x * blockDim.x + threadIdx.x;
  if (gid >= M * KP2) return;
  int m = gid / KP2, p = gid - m * KP2;
  int c = 2 * p;
  unsigned int out = 0;
  if (c < CD) {
    float2 av = *(const float2*)(A + (size_t)m * CD + c);
    float2 bv = *(const float2*)(B + (size_t)m * CD + c);
    out = pack_bf(av.x + bv.x, av.y + bv.y);
  }
  Xp[gid] = out;
}

// BatchNorm stats: two-pass column reduction (biased variance, training mode)
__global__ void bn_partial(const float* __restrict__ X, int M, int rpb,
                           float* __restrict__ partial) {
  int c = threadIdx.x;                 // blockDim = 320
  int b = blockIdx.x;
  if (c >= CD) return;
  int r0 = b * rpb;
  int r1 = r0 + rpb; if (r1 > M) r1 = M;
  float s = 0.f, q = 0.f;
  for (int r = r0; r < r1; ++r) {
    float v = X[(size_t)r * CD + c];
    s += v; q += v * v;
  }
  partial[(size_t)b * 2 * CD + c] = s;
  partial[(size_t)b * 2 * CD + CD + c] = q;
}

__global__ void bn_finalize(const float* __restrict__ partial, int NB, int M,
                            float* __restrict__ stats) {
  int c = threadIdx.x;
  if (c >= CD) return;
  float s = 0.f, q = 0.f;
  for (int b = 0; b < NB; ++b) {
    s += partial[(size_t)b * 2 * CD + c];
    q += partial[(size_t)b * 2 * CD + CD + c];
  }
  float mean = s / (float)M;
  float var = q / (float)M - mean * mean;
  stats[c] = mean;
  stats[CD + c] = rsqrtf(var + 1e-5f);
}

__global__ void bn_relu(const float* __restrict__ X, const float* __restrict__ stats,
                        const float* __restrict__ g, const float* __restrict__ b,
                        float* __restrict__ Y, long n) {
  long gid = (long)blockIdx.x * blockDim.x + threadIdx.x;
  if (gid >= n) return;
  int c = (int)(gid % CD);
  float v = (X[gid] - stats[c]) * stats[CD + c] * g[c] + b[c];
  Y[gid] = v > 0.f ? v : 0.f;
}

// relu(bn(X)) written directly as packed bf16 pairs (feeds next GEMM only)
__global__ void bn_relu_pack(const float* __restrict__ X, const float* __restrict__ stats,
                             const float* __restrict__ g, const float* __restrict__ b,
                             unsigned int* __restrict__ Xp, int M) {
  int gid = blockIdx.x * blockDim.x + threadIdx.x;
  if (gid >= M * KP2) return;
  int m = gid / KP2, p = gid - m * KP2;
  int c = 2 * p;
  unsigned int out = 0;
  if (c < CD) {
    float2 xv = *(const float2*)(X + (size_t)m * CD + c);
    float v0 = (xv.x - stats[c])   * stats[CD + c]   * g[c]   + b[c];
    float v1 = (xv.y - stats[c+1]) * stats[CD + c+1] * g[c+1] + b[c+1];
    out = pack_bf(v0 > 0.f ? v0 : 0.f, v1 > 0.f ? v1 : 0.f);
  }
  Xp[gid] = out;
}

__global__ void bn_relu_add(const float* __restrict__ X, const float* __restrict__ stats,
                            const float* __restrict__ g, const float* __restrict__ b,
                            const float* __restrict__ R, float* __restrict__ Y, long n) {
  long gid = (long)blockIdx.x * blockDim.x + threadIdx.x;
  if (gid >= n) return;
  int c = (int)(gid % CD);
  float v = (X[gid] - stats[c]) * stats[CD + c] * g[c] + b[c];
  Y[gid] = (v > 0.f ? v : 0.f) + R[gid];
}

__global__ void pool_scatter(const float* __restrict__ xcur, const int* __restrict__ batch,
                             float* __restrict__ pooled) {
  int gid = blockIdx.x * blockDim.x + threadIdx.x;
  const int total = CN * (CD / 4);
  if (gid >= total) return;
  int n = gid / (CD / 4);
  int c = (gid - n * (CD / 4)) * 4;
  int g = batch[n];
  float4 v = *(const float4*)(xcur + (size_t)n * CD + c);
  float* dp = pooled + (size_t)g * CD + c;
  atomicAdd(dp + 0, v.x);
  atomicAdd(dp + 1, v.y);
  atomicAdd(dp + 2, v.z);
  atomicAdd(dp + 3, v.w);
}

__global__ void count_nodes(const int* __restrict__ batch, float* __restrict__ cnt) {
  int n = blockIdx.x * blockDim.x + threadIdx.x;
  if (n >= CN) return;
  atomicAdd(&cnt[batch[n]], 1.0f);
}

// vnb = vn + pooled/cnt, written packed (feeds VN GEMM only)
__global__ void vn_add_pool_pack(const float* __restrict__ vn, const float* __restrict__ pooled,
                                 const float* __restrict__ cnt, unsigned int* __restrict__ Xp) {
  int gid = blockIdx.x * blockDim.x + threadIdx.x;
  if (gid >= CG * KP2) return;
  int g = gid / KP2, p = gid - g * KP2;
  int c = 2 * p;
  unsigned int out = 0;
  if (c < CD) {
    float inv = 1.0f / fmaxf(cnt[g], 1.0f);
    float2 vv = *(const float2*)(vn + (size_t)g * CD + c);
    float2 pv = *(const float2*)(pooled + (size_t)g * CD + c);
    out = pack_bf(vv.x + pv.x * inv, vv.y + pv.y * inv);
  }
  Xp[gid] = out;
}

__global__ void final_out(const float* __restrict__ pooled, const float* __restrict__ cnt,
                          const float* __restrict__ linW, const float* __restrict__ linb,
                          float* __restrict__ out) {
  __shared__ float red[128];
  int g = blockIdx.x, t = threadIdx.x;
  float inv = 1.0f / fmaxf(cnt[g], 1.0f);
  float s = 0.f;
  for (int c = t; c < CD; c += 128) s += pooled[(size_t)g * CD + c] * inv * linW[c];
  red[t] = s;
  __syncthreads();
  for (int st = 64; st > 0; st >>= 1) {
    if (t < st) red[t] += red[t + st];
    __syncthreads();
  }
  if (t == 0) out[g] = red[0] + linb[0];
}

// ---------------------------------------------------------------------------
// Launcher
// ---------------------------------------------------------------------------
extern "C" void kernel_launch(void* const* d_in, const int* in_sizes, int n_in,
                              void* d_out, int out_size, void* d_ws, size_t ws_size,
                              hipStream_t stream) {
  (void)in_sizes; (void)n_in; (void)out_size; (void)ws_size;
  const int*   x         = (const int*)d_in[0];
  const int*   edge_attr = (const int*)d_in[1];
  const int*   edge_index= (const int*)d_in[2];
  const int*   batch     = (const int*)d_in[3];
  const float* atom_emb  = (const float*)d_in[4];
  const float* bond_emb  = (const float*)d_in[5];
  const float* vn_emb    = (const float*)d_in[6];
  const float* conv_W1   = (const float*)d_in[7];
  const float* conv_b1   = (const float*)d_in[8];
  const float* conv_bng  = (const float*)d_in[9];
  const float* conv_bnb  = (const float*)d_in[10];
  const float* conv_W2   = (const float*)d_in[11];
  const float* conv_b2   = (const float*)d_in[12];
  const float* bn_g      = (const float*)d_in[13];
  const float* bn_b      = (const float*)d_in[14];
  const float* vn_W1     = (const float*)d_in[15];
  const float* vn_b1     = (const float*)d_in[16];
  const float* vn_bn1g   = (const float*)d_in[17];
  const float* vn_bn1b   = (const float*)d_in[18];
  const float* vn_W2     = (const float*)d_in[19];
  const float* vn_b2     = (const float*)d_in[20];
  const float* vn_bn2g   = (const float*)d_in[21];
  const float* vn_bn2b   = (const float*)d_in[22];
  const float* lin_W     = (const float*)d_in[23];
  const float* lin_b     = (const float*)d_in[24];
  float* out = (float*)d_out;

  // workspace carve-up
  char* base = (char*)d_ws;
  size_t off = 0;
  auto take = [&](size_t bytes) -> void* {
    void* p = base + off;
    off = (off + bytes + 255) & ~(size_t)255;
    return p;
  };
  float* xcur   = (float*)take((size_t)CN * CD * 4);
  float* ea     = (float*)take((size_t)CE * CD * 4);
  float* agg    = (float*)take((size_t)CN * CD * 4);   // also t2
  float* t1     = (float*)take((size_t)CN * CD * 4);
  float* vn     = (float*)take((size_t)CG * CD * 4);
  float* vnt    = (float*)take((size_t)CG * CD * 4);
  float* vnb    = (float*)take((size_t)CG * CD * 4);
  float* pooled = (float*)take((size_t)CG * CD * 4);
  float* cnt    = (float*)take((size_t)CG * 4);
  float* stats  = (float*)take(2 * CD * 4);
  float* partial= (float*)take((size_t)256 * 2 * CD * 4);
  unsigned int* XpN  = (unsigned int*)take((size_t)CN * KP2 * 4);
  unsigned int* XpG  = (unsigned int*)take((size_t)CG * KP2 * 4);
  unsigned int* WpC1 = (unsigned int*)take((size_t)CL * KP2 * PW * 4);
  unsigned int* WpC2 = (unsigned int*)take((size_t)CL * KP2 * PW * 4);
  unsigned int* WpV1 = (unsigned int*)take((size_t)(CL - 1) * KP2 * PW * 4);
  unsigned int* WpV2 = (unsigned int*)take((size_t)(CL - 1) * KP2 * PW * 4);
  const int MATSZ = KP2 * PW;

  auto blocks = [](long n, int bs) { return dim3((unsigned)((n + bs - 1) / bs)); };

  // --- pack all weights to bf16 padded layout ---
  pack_weights<<<blocks((long)CL * MATSZ, 256), 256, 0, stream>>>(conv_W1, WpC1, CL);
  pack_weights<<<blocks((long)CL * MATSZ, 256), 256, 0, stream>>>(conv_W2, WpC2, CL);
  pack_weights<<<blocks((long)(CL - 1) * MATSZ, 256), 256, 0, stream>>>(vn_W1, WpV1, CL - 1);
  pack_weights<<<blocks((long)(CL - 1) * MATSZ, 256), 256, 0, stream>>>(vn_W2, WpV2, CL - 1);

  // --- encoders ---
  atom_embed<<<blocks((long)CN * CD, 256), 256, 0, stream>>>(x, atom_emb, xcur);
  bond_embed<<<blocks((long)CE * CD, 256), 256, 0, stream>>>(edge_attr, bond_emb, ea);
  broadcast_vn<<<blocks((long)CG * CD, 256), 256, 0, stream>>>(vn_emb, vn);

  const long ND = (long)CN * CD;
  const long GD = (long)CG * CD;

  auto bn_stats = [&](const float* X, int M, int NB) {
    int rpb = (M + NB - 1) / NB;
    bn_partial<<<dim3(NB), dim3(320), 0, stream>>>(X, M, rpb, partial);
    bn_finalize<<<dim3(1), dim3(320), 0, stream>>>(partial, NB, M, stats);
  };
  auto gemm = [&](const unsigned int* Xp, const unsigned int* Wp, const float* bias,
                  float* Y, int M) {
    gemm_bf16_wmma<<<dim3(5, (unsigned)((M + 127) / 128)), dim3(256), 0, stream>>>(
        Xp, Wp, bias, Y, M);
  };

  for (int l = 0; l < CL; ++l) {
    // xcur += vn[batch]
    add_vn<<<blocks(ND, 256), 256, 0, stream>>>(xcur, vn, batch);
    // agg = segment_sum(relu(xcur[src] + ea), dst)
    fill_f32<<<blocks(ND, 256), 256, 0, stream>>>(agg, 0.f, ND);
    edge_scatter<<<blocks((long)CE * (CD / 4), 256), 256, 0, stream>>>(xcur, ea, edge_index, agg);
    // t0 = agg + xcur  (packed)  ->  t1 = t0 @ W1 + b1 ; relu(bn) (packed)
    add_xy_pack<<<blocks((long)CN * KP2, 256), 256, 0, stream>>>(agg, xcur, XpN, CN);
    gemm(XpN, WpC1 + (size_t)l * MATSZ, conv_b1 + l * CD, t1, CN);
    bn_stats(t1, CN, 256);
    bn_relu_pack<<<blocks((long)CN * KP2, 256), 256, 0, stream>>>(
        t1, stats, conv_bng + l * CD, conv_bnb + l * CD, XpN, CN);
    // t2 = t1b @ W2 + b2 ; relu(bn) + residual
    gemm(XpN, WpC2 + (size_t)l * MATSZ, conv_b2 + l * CD, agg, CN);
    bn_stats(agg, CN, 256);
    bn_relu_add<<<blocks(ND, 256), 256, 0, stream>>>(
        agg, stats, bn_g + l * CD, bn_b + l * CD, xcur, xcur, ND);

    if (l < CL - 1) {
      // vn = MLP(vn + mean_pool(xcur))
      fill_f32<<<blocks(GD, 256), 256, 0, stream>>>(pooled, 0.f, GD);
      fill_f32<<<blocks((long)CG, 256), 256, 0, stream>>>(cnt, 0.f, CG);
      pool_scatter<<<blocks((long)CN * (CD / 4), 256), 256, 0, stream>>>(xcur, batch, pooled);
      count_nodes<<<blocks((long)CN, 256), 256, 0, stream>>>(batch, cnt);
      vn_add_pool_pack<<<blocks((long)CG * KP2, 256), 256, 0, stream>>>(vn, pooled, cnt, XpG);
      gemm(XpG, WpV1 + (size_t)l * MATSZ, vn_b1 + l * CD, vnt, CG);
      bn_stats(vnt, CG, 32);
      bn_relu_pack<<<blocks((long)CG * KP2, 256), 256, 0, stream>>>(
          vnt, stats, vn_bn1g + l * CD, vn_bn1b + l * CD, XpG, CG);
      gemm(XpG, WpV2 + (size_t)l * MATSZ, vn_b2 + l * CD, vnb, CG);
      bn_stats(vnb, CG, 32);
      bn_relu<<<blocks(GD, 256), 256, 0, stream>>>(
          vnb, stats, vn_bn2g + l * CD, vn_bn2b + l * CD, vn, GD);
    }
  }

  // final mean-pool + linear head
  fill_f32<<<blocks(GD, 256), 256, 0, stream>>>(pooled, 0.f, GD);
  fill_f32<<<blocks((long)CG, 256), 256, 0, stream>>>(cnt, 0.f, CG);
  pool_scatter<<<blocks((long)CN * (CD / 4), 256), 256, 0, stream>>>(xcur, batch, pooled);
  count_nodes<<<blocks((long)CN, 256), 256, 0, stream>>>(batch, cnt);
  final_out<<<dim3(CG), dim3(128), 0, stream>>>(pooled, cnt, lin_W, lin_b, out);
}
